// RNNLM_35682588295550
// MI455X (gfx1250) — compile-verified, matrix operand order checked by reading, not run on hardware
//
#include <hip/hip_runtime.h>
#include <math.h>

// RNN LM: logits = scan(tanh(x@Wxh + h@Whh + b_h)) @ Whq + b_q
// Recurrence chain: fp32 (V_WMMA_F32_16X16X4_F32) to avoid compounding error.
// Logits GEMM (268 GFLOP, dominant): fp16 + V_WMMA_F32_16X16X32_F16 with
// TDM (tensor_load_to_lds) double-buffered LDS slabs + s_wait_tensorcnt.

#define VOCAB 32000
#define EMB   1024
#define HID   1024
#define TLEN  256
#define BATCH 16

#define RNN_BLOCKS 16       // workgroups cooperating on the recurrence

#define APITCH 40           // LDS row pitch in halfs (32 data + 8 pad) = 80 B
#define BPITCH 40

typedef __attribute__((ext_vector_type(2)))  float    v2f;
typedef __attribute__((ext_vector_type(4)))  float    v4f;
typedef __attribute__((ext_vector_type(8)))  float    v8f;
typedef __attribute__((ext_vector_type(16))) _Float16 v16h;
typedef __attribute__((ext_vector_type(4)))  unsigned u32x4;
typedef __attribute__((ext_vector_type(8)))  int      i32x8;
typedef __attribute__((ext_vector_type(4)))  int      i32x4;

// ---- f32 WMMA step: C += A(16x4) * B(4x16) ------------------------------
__device__ __forceinline__ v8f wmma_f32_step(v2f a, v2f b, v8f c) {
    return __builtin_amdgcn_wmma_f32_16x16x4_f32(false, a, false, b,
                                                 (short)0, c, false, false);
}

// ---- f16 WMMA step: C(f32) += A(16x32 f16) * B(32x16 f16) ---------------
__device__ __forceinline__ v8f wmma_f16_step(v16h a, v16h b, v8f c) {
    return __builtin_amdgcn_wmma_f32_16x16x32_f16(false, a, false, b,
                                                  (short)0, c, false, false);
}

// Fragment gather from an LDS row of 32 halfs (one K-chunk), ISA layout:
// VGPR0-3 = K[8*half .. +7], VGPR4-7 = K[16+8*half .. +7]
__device__ __forceinline__ v16h ld_frag16(const _Float16* p, int half) {
    union { v16h v; uint4 q[2]; } u;
    u.q[0] = *(const uint4*)(p + 8 * half);        // ds_load_b128
    u.q[1] = *(const uint4*)(p + 16 + 8 * half);   // ds_load_b128
    return u.v;
}

// ---- TDM: DMA a rows x 32-half 2D tile (row stride HID) into LDS --------
// D# per CDNA5 ISA. pad: 16 DWORDs data -> +4 DWORDs
// (row pitch 80 B in LDS == APITCH/BPITCH). Tracked by TENSORcnt.
// Toolchain here is the clang-23 6-arg builtin form.
__device__ __forceinline__ void tdm_load_2d(unsigned lds_off,
                                            const void* gaddr,
                                            unsigned rows,
                                            unsigned tensor_rows) {
    const unsigned long long ga = (unsigned long long)gaddr;
    u32x4 g0;
    g0.x = 1u;                                         // count=1, user D#
    g0.y = lds_off;                                    // lds_addr (bytes)
    g0.z = (unsigned)(ga & 0xFFFFFFFFu);               // global_addr lo
    g0.w = (unsigned)((ga >> 32) & 0x01FFFFFFu)        // global_addr hi
         | (2u << 30);                                 // type = 2 ("image")
    i32x8 g1;
    g1[0] = (1 << 16)                                  // data_size = 2 bytes
          | (1 << 20)                                  // pad_enable
          | (3 << 22)                                  // pad_interval: 16 DW
          | (3 << 25);                                 // pad_amount: 4 DW
    g1[1] = (int)((unsigned)HID << 16);                // tensor_dim0 = 1024
    g1[2] = (int)((tensor_rows & 0xFFFFu) << 16);      // tensor_dim1 lo16
    g1[3] = (int)((tensor_rows >> 16) | (32u << 16));  // dim1 hi | tile_dim0=32
    g1[4] = (int)rows;                                 // tile_dim1 (tile_dim2=0)
    g1[5] = HID;                                       // tensor_dim0_stride lo
    g1[6] = 0;                                         // stride hi | dim1_stride
    g1[7] = 0;
    const i32x4 z4 = {0, 0, 0, 0};                     // groups 2/3 unused (2D)
    const i32x8 z8 = {0, 0, 0, 0, 0, 0, 0, 0};
    __builtin_amdgcn_tensor_load_to_lds(g0, g1, z4, z4, z8, 0);
}

// ---- deterministic grid barrier (state zeroed by k_init every launch) ---
__device__ __forceinline__ void grid_sync(unsigned* counter, unsigned* gen,
                                          unsigned nblocks, unsigned target) {
    __syncthreads();
    if (threadIdx.x == 0) {
        __threadfence();
        unsigned prev = atomicAdd(counter, 1u);
        if (prev == nblocks - 1u) {
            atomicExch(counter, 0u);
            atomicAdd(gen, 1u);
        } else {
            while (atomicAdd(gen, 0u) < target) {
                __builtin_amdgcn_s_sleep(2);
            }
        }
        __threadfence();
    }
    __syncthreads();
}

__global__ void k_init(unsigned* bar) {
    bar[0] = 0u;
    bar[1] = 0u;
}

// ---- Convert W_hq [K=1024][V=32000] f32 -> BT [V][K] f16 (tiled transpose)
__global__ void k_convert_whq(const float* __restrict__ W,
                              _Float16* __restrict__ BT) {
    __shared__ float tile[32][33];
    const int v0 = blockIdx.x * 32;
    const int k0 = blockIdx.y * 32;
    const int c  = threadIdx.x & 31;
    const int r0 = threadIdx.x >> 5;
#pragma unroll
    for (int j = 0; j < 4; ++j) {
        const int r = r0 + j * 8;
        tile[r][c] = W[(long)(k0 + r) * VOCAB + v0 + c];
    }
    __syncthreads();
#pragma unroll
    for (int j = 0; j < 4; ++j) {
        const int r = r0 + j * 8;
        BT[(long)(v0 + r) * HID + k0 + c] = (_Float16)tile[c][r];
    }
}

// ---- Stage 1: pre = gather(W_e, X) @ Wxh + b_h  (f32 WMMA) --------------
__global__ void k_embed_xh(const int* __restrict__ X,
                           const float* __restrict__ We,
                           const float* __restrict__ Wxh,
                           const float* __restrict__ bh,
                           float* __restrict__ pre) {
    const int lane = threadIdx.x & 31;
    const int wave = threadIdx.x >> 5;
    const int half = lane >> 4;
    const int l15  = lane & 15;
    const int n0 = blockIdx.x * 512 + wave * 64;
    const int m0 = blockIdx.y * 16;

    const int tok = X[m0 + l15];
    const float* __restrict__ arow = We + (long)tok * EMB;

    v8f acc[4] = {};
    for (int k0 = 0; k0 < EMB; k0 += 4) {
        const int ka = k0 + 2 * half;
        v2f a;
        a.x = arow[ka + 0];
        a.y = arow[ka + 1];
        const float* b0 = Wxh + (long)ka * HID + n0 + l15;
#pragma unroll
        for (int ni = 0; ni < 4; ++ni) {
            v2f b;
            b.x = b0[ni * 16];
            b.y = b0[HID + ni * 16];
            acc[ni] = wmma_f32_step(a, b, acc[ni]);
        }
    }
#pragma unroll
    for (int ni = 0; ni < 4; ++ni) {
        const float bias = bh[n0 + ni * 16 + l15];
#pragma unroll
        for (int r = 0; r < 8; ++r) {
            const int m = m0 + r + 8 * half;
            pre[(long)m * HID + n0 + ni * 16 + l15] = acc[ni][r] + bias;
        }
    }
}

// ---- Stage 2: sequential scan (f32 WMMA), h_{t-1} staged in LDS ---------
__global__ void k_recurrence(const float* __restrict__ pre,
                             const float* __restrict__ Whh,
                             float* __restrict__ hs,
                             _Float16* __restrict__ hs16,
                             unsigned* __restrict__ bar) {
    __shared__ float hsh[BATCH * HID];             // 64 KB: full h_{t-1}
    const int lane = threadIdx.x & 31;
    const int wave = threadIdx.x >> 5;
    const int half = lane >> 4;
    const int l15  = lane & 15;
    const int n0 = (blockIdx.x * 4 + wave) * 16;

    for (int t = 0; t < TLEN; ++t) {
        v8f c = {};
        if (t > 0) {                               // uniform branch
            const v4f* __restrict__ src =
                (const v4f*)(hs + (long)(t - 1) * BATCH * HID);
            v4f* dst = (v4f*)hsh;
            for (int i = threadIdx.x; i < BATCH * HID / 4; i += blockDim.x)
                dst[i] = src[i];
            __syncthreads();

            for (int k0 = 0; k0 < HID; k0 += 4) {
                const int ka = k0 + 2 * half;
                v2f a, b;
                a.x = hsh[l15 * HID + ka + 0];
                a.y = hsh[l15 * HID + ka + 1];
                b.x = Whh[(long)(ka + 0) * HID + n0 + l15];
                b.y = Whh[(long)(ka + 1) * HID + n0 + l15];
                c = wmma_f32_step(a, b, c);
            }
            __syncthreads();
        }
        const float* __restrict__ prow = pre + (long)t * BATCH * HID;
        float* __restrict__ hrow       = hs  + (long)t * BATCH * HID;
        _Float16* __restrict__ hrow16  = hs16 + (long)t * BATCH * HID;
#pragma unroll
        for (int r = 0; r < 8; ++r) {
            const int m = r + 8 * half;
            const float hv = tanhf(c[r] + prow[m * HID + n0 + l15]);
            hrow[m * HID + n0 + l15]   = hv;
            hrow16[m * HID + n0 + l15] = (_Float16)hv;
        }
        grid_sync(bar, bar + 1, RNN_BLOCKS, (unsigned)(t + 1));
    }
}

// ---- Stage 3: logits = hs16 @ BT16^T + b_q ------------------------------
// f16 WMMA; TDM double-buffered LDS slabs (A 128x32, B 256x32, padded pitch).
// Block: 8 waves as 2(M) x 4(N); macrotile 128 x 256; wave tile 64 x 64.
__global__ void k_logits(const _Float16* __restrict__ A16,  // [4096][1024]
                         const _Float16* __restrict__ BT,   // [32000][1024]
                         const float* __restrict__ bq,
                         float* __restrict__ out) {
    __shared__ _Float16 Ash[2][128 * APITCH];      // 2 x 10 KB
    __shared__ _Float16 Bsh[2][256 * BPITCH];      // 2 x 20 KB

    const int lane = threadIdx.x & 31;
    const int wave = threadIdx.x >> 5;
    const int half = lane >> 4;
    const int l15  = lane & 15;
    const int wm = wave >> 2;                      // 0..1
    const int wn = wave & 3;                       // 0..3
    const int m0g = blockIdx.y * 128;
    const int n0g = blockIdx.x * 256;

    const unsigned ashOff[2] = { (unsigned)(unsigned long long)&Ash[0][0],
                                 (unsigned)(unsigned long long)&Ash[1][0] };
    const unsigned bshOff[2] = { (unsigned)(unsigned long long)&Bsh[0][0],
                                 (unsigned)(unsigned long long)&Bsh[1][0] };

    v8f acc[4][4] = {};

    if (wave == 0) {                               // prologue: slab for kc=0
        tdm_load_2d(ashOff[0], A16 + (long)m0g * HID, 128, TLEN * BATCH);
        tdm_load_2d(bshOff[0], BT  + (long)n0g * HID, 256, VOCAB);
    }

    for (int kc = 0; kc < HID / 32; ++kc) {
        const int cur = kc & 1;
        if (wave == 0) {
            if (kc + 1 < HID / 32) {               // issue next pair, then
                const int k1 = (kc + 1) * 32;      // wait for current pair
                tdm_load_2d(ashOff[cur ^ 1], A16 + (long)m0g * HID + k1,
                            128, TLEN * BATCH);
                tdm_load_2d(bshOff[cur ^ 1], BT + (long)n0g * HID + k1,
                            256, VOCAB);
                __builtin_amdgcn_s_wait_tensorcnt(2);  // in-order: pair kc done
            } else {
                __builtin_amdgcn_s_wait_tensorcnt(0);
            }
        }
        __syncthreads();                           // publish buf[cur]

        v16h bf[4];
#pragma unroll
        for (int ni = 0; ni < 4; ++ni)
            bf[ni] = ld_frag16(&Bsh[cur][(wn * 64 + ni * 16 + l15) * BPITCH],
                               half);
#pragma unroll
        for (int mi = 0; mi < 4; ++mi) {
            const v16h af =
                ld_frag16(&Ash[cur][(wm * 64 + mi * 16 + l15) * APITCH], half);
#pragma unroll
            for (int ni = 0; ni < 4; ++ni)
                acc[mi][ni] = wmma_f16_step(af, bf[ni], acc[mi][ni]);
        }
        __syncthreads();                           // buf[cur] free for reuse
    }

#pragma unroll
    for (int ni = 0; ni < 4; ++ni) {
        const int n = n0g + wn * 64 + ni * 16 + l15;
        const float bias = bq[n];
#pragma unroll
        for (int mi = 0; mi < 4; ++mi) {
#pragma unroll
            for (int r = 0; r < 8; ++r) {
                const int m = m0g + wm * 64 + mi * 16 + r + 8 * half;
                out[(long)m * VOCAB + n] = acc[mi][ni][r] + bias;
            }
        }
    }
}

extern "C" void kernel_launch(void* const* d_in, const int* in_sizes, int n_in,
                              void* d_out, int out_size, void* d_ws, size_t ws_size,
                              hipStream_t stream) {
    const int*   X   = (const int*)  d_in[0];   // [T,B]
    const float* We  = (const float*)d_in[1];   // [V,E]
    const float* Wxh = (const float*)d_in[2];   // [E,H]
    const float* Whh = (const float*)d_in[3];   // [H,H]
    const float* bh  = (const float*)d_in[4];   // [H]
    const float* Whq = (const float*)d_in[5];   // [H,V]
    const float* bq  = (const float*)d_in[6];   // [V]
    float* out = (float*)d_out;                 // [T,B,V]

    char* ws = (char*)d_ws;
    const size_t preB  = (size_t)TLEN * BATCH * HID * sizeof(float);     // 16 MB
    const size_t hsB   = preB;                                           // 16 MB
    const size_t h16B  = (size_t)TLEN * BATCH * HID * sizeof(_Float16);  //  8 MB
    const size_t btB   = (size_t)VOCAB * HID * sizeof(_Float16);         // 64 MB

    float*    pre  = (float*)ws;
    float*    hsb  = (float*)(ws + preB);
    _Float16* hs16 = (_Float16*)(ws + preB + hsB);
    _Float16* bt16 = (_Float16*)(ws + preB + hsB + h16B);
    unsigned* bar  = (unsigned*)(ws + preB + hsB + h16B + btB);

    k_init<<<1, 1, 0, stream>>>(bar);

    k_convert_whq<<<dim3(VOCAB / 32, HID / 32), 256, 0, stream>>>(Whq, bt16);

    k_embed_xh<<<dim3(HID / 512, (TLEN * BATCH) / 16), 256, 0, stream>>>(
        X, We, Wxh, bh, pre);

    k_recurrence<<<RNN_BLOCKS, 128, 0, stream>>>(pre, Whh, hsb, hs16, bar);

    k_logits<<<dim3(VOCAB / 256, (TLEN * BATCH) / 128), 256, 0, stream>>>(
        hs16, bt16, bq, out);
}
